// PosteriorPreference_15942918602802
// MI455X (gfx1250) — matrix operand order repaired and checked by reading, not compiled, not run on hardware
//
#include <hip/hip_runtime.h>
#include <hip/hip_bf16.h>
#include <cstdint>
#include <cstddef>

// ---------------- problem constants (from reference) ----------------
constexpr int   B_    = 64;
constexpr int   LC_   = 512;
constexpr int   LPV_  = 16;
constexpr int   LTP_  = 16;
constexpr int   V_    = 8192;
constexpr int   H_    = 1024;
constexpr int   PREF_ = 16;
constexpr int   S_    = LC_ + LPV_ + LTP_ + 1;   // 545
constexpr int   SC_   = S_ - 1;                  // 544 (copy positions)
constexpr int   BOS_  = 2;
constexpr float TAU_  = 0.67f;
constexpr float NEG_  = -1.0e9f;
constexpr float RH_   = 0.03125f;                // H^-0.5 = 1/32

typedef __bf16 bf16_t;
typedef __attribute__((ext_vector_type(16))) __bf16 v16bf;
typedef __attribute__((ext_vector_type(8)))  __bf16 v8bf;
typedef __attribute__((ext_vector_type(8)))  float  v8f;

// ---------------- block-wide reductions (256 threads, 8 wave32) ----------------
__device__ __forceinline__ float blk_max(float v, volatile float* rbuf) {
    #pragma unroll
    for (int o = 16; o; o >>= 1) v = fmaxf(v, __shfl_down(v, o));
    int lane = threadIdx.x & 31, wv = threadIdx.x >> 5;
    __syncthreads();
    if (lane == 0) rbuf[wv] = v;
    __syncthreads();
    float r = rbuf[0];
    #pragma unroll
    for (int i = 1; i < 8; i++) r = fmaxf(r, rbuf[i]);
    __syncthreads();
    return r;
}

__device__ __forceinline__ float blk_sum(float v, volatile float* rbuf) {
    #pragma unroll
    for (int o = 16; o; o >>= 1) v += __shfl_down(v, o);
    int lane = threadIdx.x & 31, wv = threadIdx.x >> 5;
    __syncthreads();
    if (lane == 0) rbuf[wv] = v;
    __syncthreads();
    float r = rbuf[0];
    #pragma unroll
    for (int i = 1; i < 8; i++) r += rbuf[i];
    __syncthreads();
    return r;
}

// ---------------- WMMA bf16 GEMM: C[M,N] = A[M,K] * Bt[N,K]^T (+bias)(tanh?) ----------------
// Bt is stored N-major so the B fragment is two contiguous 16-byte loads per
// K-chunk. Each wave accumulates MT vertically-stacked 16x16 D tiles, reusing
// the B fragment MT times (4x less B traffic on the skinny M=64 GEMMs where
// B-matrix L2 bandwidth is the binding constraint).
// grid = (N/128, M/(16*MT)), block = 256 (8 waves).
template <int MT>
__global__ __launch_bounds__(256) void k_gemm_bf16(
    const bf16_t* __restrict__ A, const bf16_t* __restrict__ Bt,
    float* __restrict__ C, bf16_t* __restrict__ Cbf,
    const float* __restrict__ bias,
    int M, int N, int K, int apply_tanh)
{
    const int wave = threadIdx.x >> 5;
    const int lane = threadIdx.x & 31;
    const int n0   = (blockIdx.x * 8 + wave) * 16;
    const int m0   = blockIdx.y * (16 * MT);
    const int kg   = lane >> 4;           // 0 or 1 (lane group)
    const int ml   = lane & 15;
    const int ncol = n0 + ml;             // B/D column for this lane

    const bf16_t* Ap[MT];
    #pragma unroll
    for (int i = 0; i < MT; i++)
        Ap[i] = A + (size_t)(m0 + i * 16 + ml) * K + kg * 8;
    const bf16_t* Bp = Bt + (size_t)ncol * K + kg * 8;

    v8f acc[MT];
    const v8f zero = {};
    #pragma unroll
    for (int i = 0; i < MT; i++) acc[i] = zero;

    for (int k0 = 0; k0 < K; k0 += 32) {
        union { v16bf v; struct { v8bf lo, hi; } p; } bu;
        bu.p.lo = *(const v8bf*)(Bp + k0);              // K = k0+kg*8    .. +7
        bu.p.hi = *(const v8bf*)(Bp + k0 + 16);         // K = k0+16+kg*8 .. +7
        const v16bf bfrag = bu.v;
        #pragma unroll
        for (int i = 0; i < MT; i++) {
            union { v16bf v; struct { v8bf lo, hi; } p; } au;
            au.p.lo = *(const v8bf*)(Ap[i] + k0);        // K = k0+kg*8    .. +7
            au.p.hi = *(const v8bf*)(Ap[i] + k0 + 16);   // K = k0+16+kg*8 .. +7
            acc[i] = __builtin_amdgcn_wmma_f32_16x16x32_bf16(
                false, au.v, false, bfrag, (short)0, acc[i], false, false);
        }
    }

    const float bv = bias ? bias[ncol] : 0.0f;
    #pragma unroll
    for (int i = 0; i < MT; i++) {
        #pragma unroll
        for (int j = 0; j < 8; j++) {
            int mm = m0 + i * 16 + j + kg * 8;
            float v = acc[i][j] + bv;
            if (apply_tanh) v = tanhf(v);
            C[(size_t)mm * N + ncol] = v;
            if (Cbf) Cbf[(size_t)mm * N + ncol] = (bf16_t)v;
        }
    }
}

// ---------------- fp32 -> bf16, transposed (for weight prep) ----------------
// W: K x N row-major  ->  Bt: N x K row-major
__global__ __launch_bounds__(256) void k_transpose_to_bf16(
    const float* __restrict__ W, bf16_t* __restrict__ Bt, int K, int N)
{
    __shared__ float tile[32][33];
    const int kb = blockIdx.x * 32, nb = blockIdx.y * 32;
    const int tx = threadIdx.x & 31, ty = threadIdx.x >> 5;
    for (int i = ty; i < 32; i += 8) {
        int k = kb + i, n = nb + tx;
        tile[i][tx] = (k < K && n < N) ? W[(size_t)k * N + n] : 0.0f;
    }
    __syncthreads();
    for (int i = ty; i < 32; i += 8) {
        int n = nb + i, k = kb + tx;
        if (n < N && k < K) Bt[(size_t)n * K + k] = (bf16_t)tile[tx][i];
    }
}

__global__ void k_convert_bf16(const float* __restrict__ X, bf16_t* __restrict__ Y, size_t n)
{
    size_t i = (size_t)blockIdx.x * blockDim.x + threadIdx.x;
    size_t st = (size_t)gridDim.x * blockDim.x;
    for (; i < n; i += st) Y[i] = (bf16_t)X[i];
}

__global__ void k_add_bf16(const float* __restrict__ a, const float* __restrict__ b,
                           bf16_t* __restrict__ y, int n)
{
    int i = blockIdx.x * 256 + threadIdx.x;
    if (i < n) y[i] = (bf16_t)(a[i] + b[i]);
}

__global__ void k_init_yprev(bf16_t* __restrict__ yprev)
{
    int i = blockIdx.x * 256 + threadIdx.x;
    if (i < B_ * V_) yprev[i] = (bf16_t)(((i % V_) == BOS_) ? 1.0f : 0.0f);
}

// ---------------- src_hidden (bf16) + src_mask + transfer_ctx ----------------
__global__ __launch_bounds__(256) void k_build_src(
    const int* __restrict__ context, const int* __restrict__ context_len,
    const float* __restrict__ pvhid, const float* __restrict__ pv_mask,
    const int* __restrict__ tp_len, const float* __restrict__ tp_hidden,
    const int* __restrict__ ar_len, const float* __restrict__ ar_hidden,
    const float* __restrict__ E_ctx, const int* __restrict__ glo2loc,
    bf16_t* __restrict__ src, float* __restrict__ smask, int* __restrict__ tctx)
{
    const int s = blockIdx.x, b = blockIdx.y;
    bf16_t* dst = src + ((size_t)b * S_ + s) * H_;
    float mk;
    if (s < LC_) {
        const int tok = context[b * LC_ + s];
        mk = (s < context_len[b]) ? 1.0f : 0.0f;
        const float* row = E_ctx + (size_t)tok * H_;
        for (int h = threadIdx.x; h < H_; h += 256) dst[h] = (bf16_t)(row[h] * mk);
        if (threadIdx.x == 0) tctx[b * LC_ + s] = glo2loc[tok];
    } else if (s < LC_ + LPV_) {
        const int j = s - LC_;
        mk = pv_mask[b * LPV_ + j];
        const float* row = pvhid + ((size_t)b * LPV_ + j) * H_;
        for (int h = threadIdx.x; h < H_; h += 256) dst[h] = (bf16_t)row[h];
    } else if (s < LC_ + LPV_ + LTP_) {
        const int j = s - (LC_ + LPV_);
        mk = (j < tp_len[b]) ? 1.0f : 0.0f;
        const float* row = tp_hidden + ((size_t)b * LTP_ + j) * H_;
        for (int h = threadIdx.x; h < H_; h += 256) dst[h] = (bf16_t)row[h];
    } else {
        mk = (ar_len[b] >= 1) ? 1.0f : 0.0f;
        const float* row = ar_hidden + (size_t)b * H_;
        for (int h = threadIdx.x; h < H_; h += 256) dst[h] = (bf16_t)row[h];
    }
    if (threadIdx.x == 0) smask[b * S_ + s] = mk;
}

// ---------------- attention: scores + softmax + a@src (one block per batch) ----------------
__global__ __launch_bounds__(256) void k_attention(
    const float* __restrict__ qk, const bf16_t* __restrict__ src,
    const float* __restrict__ smask, float* __restrict__ ctx_pre,
    bf16_t* __restrict__ ctx_pre_bf)
{
    const int b = blockIdx.x;
    __shared__ float sc[S_];
    __shared__ float rbuf[8];
    const int wave = threadIdx.x >> 5, lane = threadIdx.x & 31;
    const float* qb = qk + b * H_;

    for (int s = wave; s < S_; s += 8) {
        const bf16_t* row = src + ((size_t)b * S_ + s) * H_;
        float acc = 0.0f;
        for (int h = lane; h < H_; h += 32) acc += qb[h] * (float)row[h];
        #pragma unroll
        for (int o = 16; o; o >>= 1) acc += __shfl_down(acc, o);
        if (lane == 0) sc[s] = (smask[b * S_ + s] == 0.0f) ? NEG_ : acc * RH_;
    }
    __syncthreads();

    float m = -3.4e38f;
    for (int s = threadIdx.x; s < S_; s += 256) m = fmaxf(m, sc[s]);
    m = blk_max(m, rbuf);

    float sum = 0.0f;
    for (int s = threadIdx.x; s < S_; s += 256) {
        float e = __expf(sc[s] - m);
        sc[s] = e;
        sum += e;
    }
    sum = blk_sum(sum, rbuf);
    const float inv = 1.0f / sum;

    for (int h = threadIdx.x; h < H_; h += 256) {
        float acc = 0.0f;
        for (int s = 0; s < S_; s++) acc += sc[s] * (float)src[((size_t)b * S_ + s) * H_ + h];
        acc *= inv;
        ctx_pre[b * H_ + h] = acc;
        ctx_pre_bf[b * H_ + h] = (bf16_t)acc;
    }
}

// ---------------- copy logits: dec . src_hidden[:, :-1, :] (masked) ----------------
__global__ __launch_bounds__(256) void k_copy_logit(
    const float* __restrict__ dec, const bf16_t* __restrict__ src,
    const float* __restrict__ smask, float* __restrict__ clog)
{
    const int b = blockIdx.x;
    const int wave = threadIdx.x >> 5, lane = threadIdx.x & 31;
    const float* db = dec + b * H_;
    for (int s = wave; s < SC_; s += 8) {
        const bf16_t* row = src + ((size_t)b * S_ + s) * H_;
        float acc = 0.0f;
        for (int h = lane; h < H_; h += 32) acc += db[h] * (float)row[h];
        #pragma unroll
        for (int o = 16; o; o >>= 1) acc += __shfl_down(acc, o);
        if (lane == 0) clog[b * SC_ + s] = (smask[b * S_ + s] == 0.0f) ? NEG_ : acc;
    }
}

// ---------------- softmax over [gen(V) ; copy(544)] * H^-0.5 ----------------
__global__ __launch_bounds__(256) void k_proj_softmax(
    const float* __restrict__ glog, const float* __restrict__ clog,
    float* __restrict__ pout, float* __restrict__ cprob, int t)
{
    const int b = blockIdx.x;
    __shared__ float gl[V_];
    __shared__ float cl[SC_];
    __shared__ float rbuf[8];

    float m = -3.4e38f;
    for (int v = threadIdx.x; v < V_; v += 256) {
        float x = glog[(size_t)b * V_ + v] * RH_;
        gl[v] = x; m = fmaxf(m, x);
    }
    for (int s = threadIdx.x; s < SC_; s += 256) {
        float x = clog[b * SC_ + s] * RH_;
        cl[s] = x; m = fmaxf(m, x);
    }
    m = blk_max(m, rbuf);

    float sum = 0.0f;
    for (int v = threadIdx.x; v < V_; v += 256) { float e = __expf(gl[v] - m); gl[v] = e; sum += e; }
    for (int s = threadIdx.x; s < SC_; s += 256) { float e = __expf(cl[s] - m); cl[s] = e; sum += e; }
    sum = blk_sum(sum, rbuf);
    const float inv = 1.0f / sum;

    float* po = pout + ((size_t)b * PREF_ + t) * V_;
    for (int v = threadIdx.x; v < V_; v += 256) po[v] = gl[v] * inv;
    for (int s = threadIdx.x; s < SC_; s += 256) cprob[b * SC_ + s] = cl[s] * inv;
}

// ---------------- combine: p = gen + pv@pv_m + tp scatter + ctx scatter ----------------
__global__ __launch_bounds__(256) void k_combine(
    const float* __restrict__ cprob, const float* __restrict__ pvm,
    const int* __restrict__ tp_path, const int* __restrict__ tctx,
    float* __restrict__ pout, int t)
{
    const int b = blockIdx.x;
    __shared__ float acc[V_];
    __shared__ float cpv[LPV_];
    if (threadIdx.x < LPV_) cpv[threadIdx.x] = cprob[b * SC_ + LC_ + threadIdx.x];
    __syncthreads();

    for (int v = threadIdx.x; v < V_; v += 256) {
        float s2 = 0.0f;
        #pragma unroll
        for (int j = 0; j < LPV_; j++) s2 += cpv[j] * pvm[((size_t)b * LPV_ + j) * V_ + v];
        acc[v] = s2;
    }
    __syncthreads();

    if (threadIdx.x < LTP_) {
        int idx = tp_path[b * LTP_ + threadIdx.x];
        atomicAdd(&acc[idx], cprob[b * SC_ + LC_ + LPV_ + threadIdx.x]);
    }
    for (int l = threadIdx.x; l < LC_; l += 256)
        atomicAdd(&acc[tctx[b * LC_ + l]], cprob[b * SC_ + l]);
    __syncthreads();

    float* po = pout + ((size_t)b * PREF_ + t) * V_;
    for (int v = threadIdx.x; v < V_; v += 256) po[v] += acc[v];
}

// ---------------- gumbel softmax -> y (output + next step bf16 token dist) ----------------
__global__ __launch_bounds__(256) void k_gumbel(
    const float* __restrict__ pout, float* __restrict__ yout,
    bf16_t* __restrict__ yprev, int t)
{
    const int b = blockIdx.x;
    __shared__ float z[V_];
    __shared__ float rbuf[8];
    const float* p = pout + ((size_t)b * PREF_ + t) * V_;

    float m = -3.4e38f;
    for (int v = threadIdx.x; v < V_; v += 256) {
        unsigned long long x = (((unsigned long long)(t * B_ + b)) << 32)
                             ^ ((unsigned long long)(unsigned)v * 0x9E3779B97F4A7C15ULL);
        x ^= x >> 33; x *= 0xff51afd7ed558ccdULL;
        x ^= x >> 33; x *= 0xc4ceb9fe1a85ec53ULL;
        x ^= x >> 33;
        float u = ((float)((x >> 40) & 0xFFFFFF) + 0.5f) * (1.0f / 16777216.0f);
        u = fmaxf(u, 1e-10f);
        float g = -__logf(-__logf(u));
        float zz = (__logf(p[v] + 1e-10f) + g) * (1.0f / TAU_);
        z[v] = zz; m = fmaxf(m, zz);
    }
    m = blk_max(m, rbuf);

    float sum = 0.0f;
    for (int v = threadIdx.x; v < V_; v += 256) { float e = __expf(z[v] - m); z[v] = e; sum += e; }
    sum = blk_sum(sum, rbuf);
    const float inv = 1.0f / sum;

    float* yo = yout + ((size_t)b * PREF_ + t) * V_;
    for (int v = threadIdx.x; v < V_; v += 256) {
        float y = z[v] * inv;
        yo[v] = y;
        yprev[(size_t)b * V_ + v] = (bf16_t)y;
    }
}

// ==================================================================================
extern "C" void kernel_launch(void* const* d_in, const int* in_sizes, int n_in,
                              void* d_out, int out_size, void* d_ws, size_t ws_size,
                              hipStream_t stream)
{
    (void)in_sizes; (void)n_in; (void)out_size; (void)ws_size;

    const int*   context        = (const int*)d_in[0];
    const int*   context_len    = (const int*)d_in[1];
    const float* pv_m           = (const float*)d_in[2];
    const float* pv_m_mask      = (const float*)d_in[3];
    /* d_in[4] ar_gth unused */
    const int*   ar_gth_len     = (const int*)d_in[5];
    const int*   tp_path        = (const int*)d_in[6];
    const int*   tp_path_len    = (const int*)d_in[7];
    const float* tp_path_hidden = (const float*)d_in[8];
    const float* ar_hidden      = (const float*)d_in[9];
    const float* E_ctx          = (const float*)d_in[10];
    const float* W_p            = (const float*)d_in[11];
    const float* E_topic        = (const float*)d_in[12];
    const float* W_q            = (const float*)d_in[13];
    const float* W_k            = (const float*)d_in[14];
    const float* W_v            = (const float*)d_in[15];
    const float* W_o            = (const float*)d_in[16];
    const float* gen_W          = (const float*)d_in[17];
    const float* gen_b          = (const float*)d_in[18];
    const int*   glo2loc        = (const int*)d_in[19];

    float* out0 = (float*)d_out;                         // seq_gen_prob (B,16,V)
    float* out1 = out0 + (size_t)B_ * PREF_ * V_;        // seq[:,1:,:]  (B,16,V)

    // bump allocator over d_ws (256B aligned)
    uint8_t* w = (uint8_t*)d_ws;
    auto alloc = [&](size_t bytes) -> void* {
        void* p = (void*)w;
        w += (bytes + 255) & ~(size_t)255;
        return p;
    };
    bf16_t* srcbf = (bf16_t*)alloc((size_t)B_ * S_ * H_ * 2);   // src_hidden bf16
    bf16_t* EtT   = (bf16_t*)alloc((size_t)H_ * V_ * 2);        // E_topic^T
    bf16_t* genWT = (bf16_t*)alloc((size_t)V_ * H_ * 2);        // gen_W^T
    bf16_t* WpT   = (bf16_t*)alloc((size_t)H_ * V_ * 2);        // W_p^T
    bf16_t* WqT   = (bf16_t*)alloc((size_t)H_ * H_ * 2);
    bf16_t* Wkb   = (bf16_t*)alloc((size_t)H_ * H_ * 2);        // W_k (no transpose: qk = q @ W_k^T)
    bf16_t* WvT   = (bf16_t*)alloc((size_t)H_ * H_ * 2);
    bf16_t* WoT   = (bf16_t*)alloc((size_t)H_ * H_ * 2);
    bf16_t* pvbf  = (bf16_t*)alloc((size_t)B_ * LPV_ * V_ * 2);
    float*  pvhid = (float*)alloc((size_t)B_ * LPV_ * H_ * 4);
    float*  smask = (float*)alloc((size_t)B_ * S_ * 4);
    int*    tctx  = (int*)alloc((size_t)B_ * LC_ * 4);
    bf16_t* yprev = (bf16_t*)alloc((size_t)B_ * V_ * 2);
    float*  embf  = (float*)alloc((size_t)B_ * H_ * 4);
    bf16_t* embb  = (bf16_t*)alloc((size_t)B_ * H_ * 2);
    float*  qf    = (float*)alloc((size_t)B_ * H_ * 4);
    bf16_t* qb    = (bf16_t*)alloc((size_t)B_ * H_ * 2);
    float*  qkf   = (float*)alloc((size_t)B_ * H_ * 4);
    float*  cpre  = (float*)alloc((size_t)B_ * H_ * 4);
    bf16_t* cpreb = (bf16_t*)alloc((size_t)B_ * H_ * 2);
    float*  ctxf  = (float*)alloc((size_t)B_ * H_ * 4);
    bf16_t* aob   = (bf16_t*)alloc((size_t)B_ * H_ * 2);
    float*  decf  = (float*)alloc((size_t)B_ * H_ * 4);
    bf16_t* decb  = (bf16_t*)alloc((size_t)B_ * H_ * 2);
    float*  glog  = (float*)alloc((size_t)B_ * V_ * 4);
    float*  clog  = (float*)alloc((size_t)B_ * SC_ * 4);
    float*  cprob = (float*)alloc((size_t)B_ * SC_ * 4);

    const dim3 tb(256);

    // ---- precompute: bf16 weight prep (transposed to N-major for contiguous B frags) ----
    k_transpose_to_bf16<<<dim3(V_ / 32, H_ / 32), tb, 0, stream>>>(E_topic, EtT, V_, H_);
    k_transpose_to_bf16<<<dim3(H_ / 32, V_ / 32), tb, 0, stream>>>(gen_W, genWT, H_, V_);
    k_transpose_to_bf16<<<dim3(V_ / 32, H_ / 32), tb, 0, stream>>>(W_p, WpT, V_, H_);
    k_transpose_to_bf16<<<dim3(H_ / 32, H_ / 32), tb, 0, stream>>>(W_q, WqT, H_, H_);
    k_convert_bf16<<<1024, tb, 0, stream>>>(W_k, Wkb, (size_t)H_ * H_);
    k_transpose_to_bf16<<<dim3(H_ / 32, H_ / 32), tb, 0, stream>>>(W_v, WvT, H_, H_);
    k_transpose_to_bf16<<<dim3(H_ / 32, H_ / 32), tb, 0, stream>>>(W_o, WoT, H_, H_);
    k_convert_bf16<<<2048, tb, 0, stream>>>(pv_m, pvbf, (size_t)B_ * LPV_ * V_);

    // pv_hidden = pv_m @ W_p : M=1024, N=1024, K=8192 (WMMA, 4 M-tiles/wave)
    k_gemm_bf16<4><<<dim3(H_ / 128, (B_ * LPV_) / 64), tb, 0, stream>>>(
        pvbf, WpT, pvhid, nullptr, nullptr, B_ * LPV_, H_, V_, 0);

    k_build_src<<<dim3(S_, B_), tb, 0, stream>>>(
        context, context_len, pvhid, pv_m_mask, tp_path_len, tp_path_hidden,
        ar_gth_len, ar_hidden, E_ctx, glo2loc, srcbf, smask, tctx);

    k_init_yprev<<<(B_ * V_) / 256, tb, 0, stream>>>(yprev);

    // ---- 16 decode steps ----
    for (int t = 0; t < PREF_; t++) {
        // emb = y_prev @ E_topic        (M=64, N=1024, K=8192)
        k_gemm_bf16<4><<<dim3(H_ / 128, B_ / 64), tb, 0, stream>>>(
            yprev, EtT, embf, embb, nullptr, B_, H_, V_, 0);
        // q = emb @ W_q                 (M=64, N=1024, K=1024)
        k_gemm_bf16<4><<<dim3(H_ / 128, B_ / 64), tb, 0, stream>>>(
            embb, WqT, qf, qb, nullptr, B_, H_, H_, 0);
        // qk = q @ W_k^T  (folds W_k into query so K matrix is never materialized)
        k_gemm_bf16<4><<<dim3(H_ / 128, B_ / 64), tb, 0, stream>>>(
            qb, Wkb, qkf, nullptr, nullptr, B_, H_, H_, 0);
        // attention scores/softmax + ctx_pre = a @ src_hidden
        k_attention<<<B_, tb, 0, stream>>>(qkf, srcbf, smask, cpre, cpreb);
        // ctx = ctx_pre @ W_v
        k_gemm_bf16<4><<<dim3(H_ / 128, B_ / 64), tb, 0, stream>>>(
            cpreb, WvT, ctxf, nullptr, nullptr, B_, H_, H_, 0);
        // ao = emb + ctx
        k_add_bf16<<<(B_ * H_) / 256, tb, 0, stream>>>(embf, ctxf, aob, B_ * H_);
        // dec = tanh(ao @ W_o)
        k_gemm_bf16<4><<<dim3(H_ / 128, B_ / 64), tb, 0, stream>>>(
            aob, WoT, decf, decb, nullptr, B_, H_, H_, 1);
        // gen_logit = dec @ gen_W + gen_b   (M=64, N=8192, K=1024)
        k_gemm_bf16<4><<<dim3(V_ / 128, B_ / 64), tb, 0, stream>>>(
            decb, genWT, glog, nullptr, gen_b, B_, V_, H_, 0);
        // copy_logit = dec . src_hidden[:, :-1, :] (masked)
        k_copy_logit<<<B_, tb, 0, stream>>>(decf, srcbf, smask, clog);
        // softmax over [gen ; copy] -> gen_prob (into out0) + copy probs
        k_proj_softmax<<<B_, tb, 0, stream>>>(glog, clog, out0, cprob, t);
        // p += pv@pv_m + tp scatter + ctx scatter (LDS accumulate)
        k_combine<<<B_, tb, 0, stream>>>(cprob, pv_m, tp_path, tctx, out0, t);
        // y = gumbel-softmax(p) -> out1 + next-step token dist (bf16)
        k_gumbel<<<B_, tb, 0, stream>>>(out0, out1, yprev, t);
    }
}
